// CausalSelfAttention2D_44049184587836
// MI455X (gfx1250) — compile-verified, matrix operand order checked by reading, not run on hardware
//
#include <hip/hip_runtime.h>
#include <hip/hip_bf16.h>

// ---------------------------------------------------------------------------
// CausalSelfAttention2D for MI455X (gfx1250)
//   wave32 + v_wmma_f32_16x16x32_bf16 + Tensor Data Mover (async LDS staging)
// B=8, C=512, H=W=32 -> N=1024, NH=8, HD=64
// ---------------------------------------------------------------------------

#define BATCH 8
#define CCH   512
#define SEQ   1024
#define NH    8
#define HD    64

typedef __bf16 bf16_t;
typedef __attribute__((ext_vector_type(16))) __bf16 v16bf;
typedef __attribute__((ext_vector_type(8)))  __bf16 v8bf;
typedef __attribute__((ext_vector_type(8)))  float  v8f;
typedef __attribute__((ext_vector_type(4)))  unsigned int u32x4;
typedef __attribute__((ext_vector_type(8)))  int i32x8;
typedef __attribute__((ext_vector_type(4)))  int i32x4;

union V16 { v16bf v; v8bf h[2]; bf16_t e[16]; };
union V8F { v8f  v; float  e[8]; };

__device__ __forceinline__ bf16_t f2bf(float f) { return (bf16_t)f; }

#define WMMA_BF16(A,B,C) \
  __builtin_amdgcn_wmma_f32_16x16x32_bf16(false,(A),false,(B),(short)0,(C),false,false)

// ---------------------------------------------------------------------------
// TDM: async 2D tile load (global -> LDS), bf16 elements (data_size code 1).
// Descriptor built per CDNA5 ISA 8.3/8.4 (D# group0 128b + group1 256b).
// Tracked by TENSORcnt; in-order per wave.
// ---------------------------------------------------------------------------
__device__ __forceinline__ void tdm_load_2d(unsigned lds_addr, const void* gptr,
                                            unsigned tile_d0, unsigned tile_d1,
                                            unsigned tensor_d0, unsigned tensor_d1,
                                            unsigned long long stride0) {
  const unsigned long long ga = (unsigned long long)(size_t)gptr;
  u32x4 g0;
  g0[0] = 1u;                                        // count=1, user mode
  g0[1] = lds_addr;                                  // LDS byte address
  g0[2] = (unsigned)(ga & 0xffffffffu);              // global_addr[31:0]
  g0[3] = (unsigned)((ga >> 32) & 0x01ffffffu) | (2u << 30);  // addr[56:32], type=2
  i32x8 g1;
  g1[0] = (int)(1u << 16);                           // wg_mask=0, data_size=1 (2B)
  g1[1] = (int)((tensor_d0 & 0xffffu) << 16);        // tensor_dim0[15:0]
  g1[2] = (int)(((tensor_d0 >> 16) & 0xffffu) | ((tensor_d1 & 0xffffu) << 16));
  g1[3] = (int)(((tensor_d1 >> 16) & 0xffffu) | ((tile_d0 & 0xffffu) << 16));
  g1[4] = (int)(tile_d1 & 0xffffu);                  // tile_dim1, tile_dim2=0
  g1[5] = (int)(stride0 & 0xffffffffu);              // tensor_dim0_stride[31:0]
  g1[6] = (int)((stride0 >> 32) & 0xffffu);          // stride0[47:32], stride1=0
  g1[7] = 0;
  const i32x4 z4 = {0, 0, 0, 0};
#if __clang_major__ >= 23
  const i32x8 z8 = {0, 0, 0, 0, 0, 0, 0, 0};
  __builtin_amdgcn_tensor_load_to_lds(g0, g1, z4, z4, z8, 0);
#else
  __builtin_amdgcn_tensor_load_to_lds(g0, g1, z4, z4, 0);
#endif
}

// ---------------------------------------------------------------------------
// Kernel 1: qkv = x_seq @ W_qkv^T + b_qkv   (8192 x 1536), K = 512
// 32(M) x 64(N) tile per wave: 2 A-tiles, 8 accumulators, 8 WMMA per K-step.
// Q scaled by hd^-0.5 at store; Q,K stored [b,h,n,d]; V stored transposed [b,h,d,n].
// ---------------------------------------------------------------------------
__global__ void __launch_bounds__(256)
qkv_wmma_kernel(const float* __restrict__ x, const float* __restrict__ wqkv,
                const float* __restrict__ bqkv,
                bf16_t* __restrict__ qg, bf16_t* __restrict__ kg,
                bf16_t* __restrict__ vtg) {
  const int lane = threadIdx.x & 31;
  const int wid  = threadIdx.x >> 5;
  const int hi   = lane >> 4;
  const int ln   = lane & 15;
  const int gw   = blockIdx.x * 8 + wid;   // 0..6143
  const int rt   = gw / 24;                // 32-row tile of 8192
  const int cg   = gw % 24;                // 64-col group of 1536

  const int row0 = rt * 32 + ln;
  const int row1 = row0 + 16;
  const float* xcol0 = x + (size_t)(row0 >> 10) * CCH * SEQ + (row0 & 1023);
  const float* xcol1 = x + (size_t)(row1 >> 10) * CCH * SEQ + (row1 & 1023);

  V8F acc[2][4];
#pragma unroll
  for (int m = 0; m < 2; ++m)
#pragma unroll
    for (int t = 0; t < 4; ++t)
#pragma unroll
      for (int r = 0; r < 8; ++r) acc[m][t].e[r] = 0.0f;

  for (int kb = 0; kb < CCH; kb += 32) {
    if (kb + 32 < CCH) {                    // global_prefetch_b8
      __builtin_prefetch(xcol0 + (size_t)(kb + 32 + hi * 8) * SEQ, 0, 1);
      __builtin_prefetch(xcol1 + (size_t)(kb + 32 + hi * 8) * SEQ, 0, 1);
    }
    // A tiles (16x32 bf16): lane ln = row M, K runs [hi*8,+8) and [16+hi*8,+8)
    V16 a0, a1;
#pragma unroll
    for (int e = 0; e < 8; ++e) {
      a0.e[e]     = f2bf(xcol0[(size_t)(kb + hi * 8 + e) * SEQ]);
      a0.e[8 + e] = f2bf(xcol0[(size_t)(kb + 16 + hi * 8 + e) * SEQ]);
      a1.e[e]     = f2bf(xcol1[(size_t)(kb + hi * 8 + e) * SEQ]);
      a1.e[8 + e] = f2bf(xcol1[(size_t)(kb + 16 + hi * 8 + e) * SEQ]);
    }
#pragma unroll
    for (int t = 0; t < 4; ++t) {
      const int col = cg * 64 + t * 16 + ln;   // B lane = output column N
      const float* wr = wqkv + (size_t)col * CCH + kb + hi * 8;
      float4 f0 = ((const float4*)wr)[0];
      float4 f1 = ((const float4*)wr)[1];
      float4 f2 = ((const float4*)(wr + 16))[0];
      float4 f3 = ((const float4*)(wr + 16))[1];
      V16 bm;
      bm.e[0]=f2bf(f0.x);  bm.e[1]=f2bf(f0.y);  bm.e[2]=f2bf(f0.z);  bm.e[3]=f2bf(f0.w);
      bm.e[4]=f2bf(f1.x);  bm.e[5]=f2bf(f1.y);  bm.e[6]=f2bf(f1.z);  bm.e[7]=f2bf(f1.w);
      bm.e[8]=f2bf(f2.x);  bm.e[9]=f2bf(f2.y);  bm.e[10]=f2bf(f2.z); bm.e[11]=f2bf(f2.w);
      bm.e[12]=f2bf(f3.x); bm.e[13]=f2bf(f3.y); bm.e[14]=f2bf(f3.z); bm.e[15]=f2bf(f3.w);
      acc[0][t].v = WMMA_BF16(a0.v, bm.v, acc[0][t].v);
      acc[1][t].v = WMMA_BF16(a1.v, bm.v, acc[1][t].v);
    }
  }

  // Epilogue: bias, split into Q (scaled) / K / V^T
#pragma unroll
  for (int t = 0; t < 4; ++t) {
    const int col   = cg * 64 + t * 16 + ln;
    const float bias = bqkv[col];
    const int which = col >> 9;       // 0=Q 1=K 2=V
    const int c     = col & 511;
    const int h     = c >> 6;
    const int hd    = c & 63;
#pragma unroll
    for (int m = 0; m < 2; ++m)
#pragma unroll
      for (int r = 0; r < 8; ++r) {
        const int rrow = rt * 32 + m * 16 + r + hi * 8;
        const int b2 = rrow >> 10;
        const int n2 = rrow & 1023;
        const size_t bh = (size_t)(b2 * NH + h);
        const float val = acc[m][t].e[r] + bias;
        if (which == 0)      qg[(bh * SEQ + n2) * HD + hd]  = f2bf(val * 0.125f);
        else if (which == 1) kg[(bh * SEQ + n2) * HD + hd]  = f2bf(val);
        else                 vtg[(bh * HD + hd) * SEQ + n2] = f2bf(val);
      }
  }
}

// ---------------------------------------------------------------------------
// Kernel 2: flash-style causal attention, one wave per (b, head, 16-query tile).
// K and V^T 32-key blocks are DMA'd into double-buffered LDS by the TDM
// (tensor_load_to_lds, TENSORcnt), overlapping DMA of block j+1 with compute
// of block j. Online softmax; P relaid D->A via LDS.
// ---------------------------------------------------------------------------
#define KV_TILE 2048   // 32 x 64 bf16 elements = 4 KB

__global__ void __launch_bounds__(64)
attn_wmma_kernel(const bf16_t* __restrict__ qg, const bf16_t* __restrict__ kg,
                 const bf16_t* __restrict__ vtg, bf16_t* __restrict__ att) {
  __shared__ bf16_t kbuf[2][2][KV_TILE];   // [wave][buf][32 rows x 64 d]
  __shared__ bf16_t vbuf[2][2][KV_TILE];   // [wave][buf][64 d x 32 j]
  __shared__ bf16_t pstage[2][16 * 32];    // per-wave P staging

  const int lane = threadIdx.x & 31;
  const int wid  = threadIdx.x >> 5;       // 0..1
  const int hi   = lane >> 4;
  const int ln   = lane & 15;
  const int gw   = blockIdx.x * 2 + wid;   // 0..4095
  const int qt   = gw & 63;                // query tile 0..63
  const int bh   = gw >> 6;                // b*8 + head
  const int qbase = qt * 16;

  const bf16_t* kbase = kg  + (size_t)bh * SEQ * HD;
  const bf16_t* vbase = vtg + (size_t)bh * HD * SEQ;

  // Q as two 16x32 A tiles (d = 0..31 and 32..63), pre-scaled by hd^-0.5
  const bf16_t* qrow = qg + ((size_t)bh * SEQ + qbase + ln) * HD;
  V16 qa0, qa1;
  qa0.h[0] = *(const v8bf*)(qrow + 0  + hi * 8);
  qa0.h[1] = *(const v8bf*)(qrow + 16 + hi * 8);
  qa1.h[0] = *(const v8bf*)(qrow + 32 + hi * 8);
  qa1.h[1] = *(const v8bf*)(qrow + 48 + hi * 8);

  float m[8], l[8];
  V8F acc[4];
#pragma unroll
  for (int r = 0; r < 8; ++r) { m[r] = -1e30f; l[r] = 0.0f; }
#pragma unroll
  for (int t = 0; t < 4; ++t)
#pragma unroll
    for (int r = 0; r < 8; ++r) acc[t].e[r] = 0.0f;

  const int njb = qt / 2 + 1;              // 32-key blocks needed (causal)

  // Prime buffer 0: K tile (32 rows x 64 d), V^T tile (64 d x 32 j)
  tdm_load_2d((unsigned)(size_t)&kbuf[wid][0][0], kbase, HD, 32, HD, SEQ, HD);
  tdm_load_2d((unsigned)(size_t)&vbuf[wid][0][0], vbase, 32, HD, SEQ, HD, SEQ);

  for (int jb = 0; jb < njb; ++jb) {
    const int jbase = jb * 32;
    const int cur = jb & 1;

    if (jb + 1 < njb) {
      const int nxt = cur ^ 1;
      const int jn = jbase + 32;
      tdm_load_2d((unsigned)(size_t)&kbuf[wid][nxt][0], kbase + (size_t)jn * HD,
                  HD, 32, HD, SEQ, HD);
      tdm_load_2d((unsigned)(size_t)&vbuf[wid][nxt][0], vbase + jn,
                  32, HD, SEQ, HD, SEQ);
      __builtin_amdgcn_s_wait_tensorcnt(2);   // current buffer's 2 loads landed
    } else {
      __builtin_amdgcn_s_wait_tensorcnt(0);
    }

    const bf16_t* kt = kbuf[wid][cur];
    const bf16_t* vt = vbuf[wid][cur];

    // S = Q K^T : K B-tiles from LDS; lane ln = key column j
    const bf16_t* krow0 = kt + (size_t)ln * HD;          // j_local = ln
    const bf16_t* krow1 = kt + (size_t)(16 + ln) * HD;   // j_local = 16+ln
    V16 kb00, kb01, kb10, kb11;
    kb00.h[0] = *(const v8bf*)(krow0 + 0  + hi * 8);
    kb00.h[1] = *(const v8bf*)(krow0 + 16 + hi * 8);
    kb01.h[0] = *(const v8bf*)(krow0 + 32 + hi * 8);
    kb01.h[1] = *(const v8bf*)(krow0 + 48 + hi * 8);
    kb10.h[0] = *(const v8bf*)(krow1 + 0  + hi * 8);
    kb10.h[1] = *(const v8bf*)(krow1 + 16 + hi * 8);
    kb11.h[0] = *(const v8bf*)(krow1 + 32 + hi * 8);
    kb11.h[1] = *(const v8bf*)(krow1 + 48 + hi * 8);

    V8F s0, s1;
#pragma unroll
    for (int r = 0; r < 8; ++r) { s0.e[r] = 0.0f; s1.e[r] = 0.0f; }
    s0.v = WMMA_BF16(qa0.v, kb00.v, s0.v);
    s0.v = WMMA_BF16(qa1.v, kb01.v, s0.v);
    s1.v = WMMA_BF16(qa0.v, kb10.v, s1.v);
    s1.v = WMMA_BF16(qa1.v, kb11.v, s1.v);

    // Causal mask: only the last block can straddle the diagonal
    if (jb == njb - 1) {
#pragma unroll
      for (int r = 0; r < 8; ++r) {
        const int i  = qbase + r + hi * 8;
        if (jbase + ln > i)      s0.e[r] = -1e30f;
        if (jbase + 16 + ln > i) s1.e[r] = -1e30f;
      }
    }

    // Online softmax (row stats via 16-lane-half shfl reductions)
#pragma unroll
    for (int r = 0; r < 8; ++r) {
      float mt = fmaxf(s0.e[r], s1.e[r]);
#pragma unroll
      for (int off = 1; off < 16; off <<= 1)
        mt = fmaxf(mt, __shfl_xor(mt, off, 32));
      const float mn = fmaxf(m[r], mt);
      const float sc = __expf(m[r] - mn);
      m[r] = mn;
      const float p0 = __expf(s0.e[r] - mn);
      const float p1 = __expf(s1.e[r] - mn);
      s0.e[r] = p0; s1.e[r] = p1;
      float rs = p0 + p1;
#pragma unroll
      for (int off = 1; off < 16; off <<= 1)
        rs += __shfl_xor(rs, off, 32);
      l[r] = l[r] * sc + rs;
#pragma unroll
      for (int t = 0; t < 4; ++t) acc[t].e[r] *= sc;
    }

    // P: D-layout -> A-layout through LDS (same-wave DS ops stay in order)
    bf16_t* ps = pstage[wid];
#pragma unroll
    for (int r = 0; r < 8; ++r) {
      const int row = r + hi * 8;
      ps[row * 32 + ln]      = f2bf(s0.e[r]);
      ps[row * 32 + 16 + ln] = f2bf(s1.e[r]);
    }
    V16 pa;
    const bf16_t* prow = ps + ln * 32;
    pa.h[0] = *(const v8bf*)(prow + hi * 8);
    pa.h[1] = *(const v8bf*)(prow + 16 + hi * 8);

    // out += P @ V : V^T rows in LDS give contiguous B-tile loads, 4 d-tiles
#pragma unroll
    for (int t = 0; t < 4; ++t) {
      const bf16_t* vrow = vt + (size_t)(t * 16 + ln) * 32;
      V16 vb;
      vb.h[0] = *(const v8bf*)(vrow + hi * 8);
      vb.h[1] = *(const v8bf*)(vrow + 16 + hi * 8);
      acc[t].v = WMMA_BF16(pa.v, vb.v, acc[t].v);
    }
  }

  // Normalize and store attention output as [b, n, c] bf16 for the proj GEMM
  const int b2 = bh >> 3;
  const int h2 = bh & 7;
#pragma unroll
  for (int t = 0; t < 4; ++t)
#pragma unroll
    for (int r = 0; r < 8; ++r) {
      const int i = qbase + r + hi * 8;
      const int d = t * 16 + ln;
      const float o = acc[t].e[r] / l[r];
      att[((size_t)(b2 * SEQ + i)) * CCH + h2 * HD + d] = f2bf(o);
    }
}

// ---------------------------------------------------------------------------
// Kernel 3: out = att @ W_proj^T + b_proj, scattered back to NCHW fp32.
// 32(M) x 64(N) tile per wave, A rows contiguous bf16.
// ---------------------------------------------------------------------------
__global__ void __launch_bounds__(256)
proj_wmma_kernel(const bf16_t* __restrict__ att, const float* __restrict__ wproj,
                 const float* __restrict__ bproj, float* __restrict__ out) {
  const int lane = threadIdx.x & 31;
  const int wid  = threadIdx.x >> 5;
  const int hi   = lane >> 4;
  const int ln   = lane & 15;
  const int gw   = blockIdx.x * 8 + wid;   // 0..2047
  const int rt   = gw / 8;                 // 32-row tile
  const int cg   = gw % 8;                 // 64-col group

  const bf16_t* arow0 = att + (size_t)(rt * 32 + ln) * CCH;
  const bf16_t* arow1 = arow0 + (size_t)16 * CCH;

  V8F acc[2][4];
#pragma unroll
  for (int m = 0; m < 2; ++m)
#pragma unroll
    for (int t = 0; t < 4; ++t)
#pragma unroll
      for (int r = 0; r < 8; ++r) acc[m][t].e[r] = 0.0f;

  for (int kb = 0; kb < CCH; kb += 32) {
    V16 a0, a1;
    a0.h[0] = *(const v8bf*)(arow0 + kb + hi * 8);
    a0.h[1] = *(const v8bf*)(arow0 + kb + 16 + hi * 8);
    a1.h[0] = *(const v8bf*)(arow1 + kb + hi * 8);
    a1.h[1] = *(const v8bf*)(arow1 + kb + 16 + hi * 8);
#pragma unroll
    for (int t = 0; t < 4; ++t) {
      const int col = cg * 64 + t * 16 + ln;
      const float* wr = wproj + (size_t)col * CCH + kb + hi * 8;
      float4 f0 = ((const float4*)wr)[0];
      float4 f1 = ((const float4*)wr)[1];
      float4 f2 = ((const float4*)(wr + 16))[0];
      float4 f3 = ((const float4*)(wr + 16))[1];
      V16 bm;
      bm.e[0]=f2bf(f0.x);  bm.e[1]=f2bf(f0.y);  bm.e[2]=f2bf(f0.z);  bm.e[3]=f2bf(f0.w);
      bm.e[4]=f2bf(f1.x);  bm.e[5]=f2bf(f1.y);  bm.e[6]=f2bf(f1.z);  bm.e[7]=f2bf(f1.w);
      bm.e[8]=f2bf(f2.x);  bm.e[9]=f2bf(f2.y);  bm.e[10]=f2bf(f2.z); bm.e[11]=f2bf(f2.w);
      bm.e[12]=f2bf(f3.x); bm.e[13]=f2bf(f3.y); bm.e[14]=f2bf(f3.z); bm.e[15]=f2bf(f3.w);
      acc[0][t].v = WMMA_BF16(a0.v, bm.v, acc[0][t].v);
      acc[1][t].v = WMMA_BF16(a1.v, bm.v, acc[1][t].v);
    }
  }

#pragma unroll
  for (int t = 0; t < 4; ++t) {
    const int col = cg * 64 + t * 16 + ln;
    const float bias = bproj[col];
#pragma unroll
    for (int m = 0; m < 2; ++m)
#pragma unroll
      for (int r = 0; r < 8; ++r) {
        const int rrow = rt * 32 + m * 16 + r + hi * 8;
        const int b2 = rrow >> 10;
        const int n2 = rrow & 1023;
        out[((size_t)b2 * CCH + col) * SEQ + n2] = acc[m][t].e[r] + bias;
      }
  }
}

// ---------------------------------------------------------------------------
extern "C" void kernel_launch(void* const* d_in, const int* in_sizes, int n_in,
                              void* d_out, int out_size, void* d_ws, size_t ws_size,
                              hipStream_t stream) {
  const float* x     = (const float*)d_in[0];
  const float* wqkv  = (const float*)d_in[1];
  const float* bqkv  = (const float*)d_in[2];
  const float* wproj = (const float*)d_in[3];
  const float* bproj = (const float*)d_in[4];
  float* out = (float*)d_out;

  const size_t QE = (size_t)BATCH * NH * SEQ * HD;   // 4M bf16 elems each
  bf16_t* qg  = (bf16_t*)d_ws;
  bf16_t* kg  = qg + QE;
  bf16_t* vtg = kg + QE;
  bf16_t* att = vtg + QE;                            // 8192 x 512 bf16

  qkv_wmma_kernel <<< 768, 256, 0, stream>>>(x, wqkv, bqkv, qg, kg, vtg);
  attn_wmma_kernel<<<2048,  64, 0, stream>>>(qg, kg, vtg, att);
  proj_wmma_kernel<<< 256, 256, 0, stream>>>(att, wproj, bproj, out);
}
